// LSTMCell_58016418234581
// MI455X (gfx1250) — matrix operand order repaired and checked by reading, not compiled, not run on hardware
//
#include <hip/hip_runtime.h>

#define IN_DIM   512
#define HID_DIM  512
#define BATCH    16384
#define K_TOTAL  (IN_DIM + HID_DIM)     // 1024
#define BM       64
#define BN       32
#define KC       64
#define LDK      72                      // padded K stride in bf16 elems (144B row pitch)
#define NSTAGES  (K_TOTAL / KC)          // 16

typedef __attribute__((ext_vector_type(16))) __bf16 v16bf;
typedef __attribute__((ext_vector_type(8)))  float  v8f;

union Frag { v16bf v; uint4 q[2]; };

// ---------------- f32 -> packed bf16 (1 VALU op per pair where possible) -----------
__device__ __forceinline__ unsigned int cvt2_bf16(float lo, float hi) {
#if __has_builtin(__builtin_amdgcn_cvt_pk_bf16_f32)
    typedef __attribute__((ext_vector_type(2))) __bf16 v2bf;
    union { v2bf v; unsigned int u; } r;
    r.v = __builtin_amdgcn_cvt_pk_bf16_f32(lo, hi);
    return r.u;
#else
    unsigned int a = __float_as_uint(lo) + 0x8000u;   // round-to-nearest
    unsigned int b = __float_as_uint(hi) + 0x8000u;
    return __builtin_amdgcn_perm(b, a, 0x07060302u);  // {b[31:16], a[31:16]}
#endif
}

__device__ __forceinline__ float sigf(float v)     { return 1.0f / (1.0f + __expf(-v)); }
__device__ __forceinline__ float tanhfast(float v) { return 2.0f / (1.0f + __expf(-2.0f * v)) - 1.0f; }

// ---------------- CDNA5 async global->LDS copy (ASYNCcnt-tracked, no VGPR data) ----
__device__ __forceinline__ void async_b128(unsigned int lds_off, const void* gptr) {
    asm volatile("global_load_async_to_lds_b128 %0, %1, off"
                 :: "v"(lds_off), "v"(gptr) : "memory");
}
__device__ __forceinline__ void wait_async0() {
    asm volatile("s_wait_asynccnt 0x0" ::: "memory");
}

// =========================== Phase 1: convert to bf16 ==============================
__global__ __launch_bounds__(256)
void convert_A_kernel(const float* __restrict__ x, const float* __restrict__ h,
                      unsigned short* __restrict__ Abuf) {
    const size_t t   = (size_t)blockIdx.x * 256 + threadIdx.x;
    const int row    = (int)(t >> 7);          // 0..16383
    const int kseg   = (int)(t & 127) << 3;    // 0..1016 step 8
    const float* src = (kseg < IN_DIM)
        ? x + (size_t)row * IN_DIM  + kseg
        : h + (size_t)row * HID_DIM + (kseg - IN_DIM);
    float4 v0 = ((const float4*)src)[0];
    float4 v1 = ((const float4*)src)[1];
    uint4 o;
    o.x = cvt2_bf16(v0.x, v0.y); o.y = cvt2_bf16(v0.z, v0.w);
    o.z = cvt2_bf16(v1.x, v1.y); o.w = cvt2_bf16(v1.z, v1.w);
    *(uint4*)(Abuf + (size_t)row * K_TOTAL + kseg) = o;
}

struct WParams {
    const float* Wx[4];    // gate order i, g, f, o
    const float* Wh[4];
    unsigned short* out;   // [4*HID][K_TOTAL] bf16
};

__global__ __launch_bounds__(256)
void convert_W_kernel(WParams wp) {
    const size_t t  = (size_t)blockIdx.x * 256 + threadIdx.x;
    const int row   = (int)(t >> 7);           // 0..2047 = g*512 + j
    const int kseg  = (int)(t & 127) << 3;
    const int g     = row >> 9;
    const int j     = row & (HID_DIM - 1);
    const float* src = (kseg < IN_DIM)
        ? wp.Wx[g] + (size_t)j * IN_DIM  + kseg
        : wp.Wh[g] + (size_t)j * HID_DIM + (kseg - IN_DIM);
    float4 v0 = ((const float4*)src)[0];
    float4 v1 = ((const float4*)src)[1];
    uint4 o;
    o.x = cvt2_bf16(v0.x, v0.y); o.y = cvt2_bf16(v0.z, v0.w);
    o.z = cvt2_bf16(v1.x, v1.y); o.w = cvt2_bf16(v1.z, v1.w);
    *(uint4*)(wp.out + (size_t)row * K_TOTAL + kseg) = o;
}

// ======================= Phase 2: WMMA GEMM + fused LSTM ===========================
struct GParams {
    const unsigned short* A;   // [BATCH][1024] bf16   (x | h)
    const unsigned short* W;   // [2048][1024] bf16    (gate-major i,g,f,o; Wx | Wh)
    const float* c;
    const float* bx[4];
    const float* bh[4];
    float* h_out;
    float* c_out;
};

__global__ __launch_bounds__(256)
void lstm_wmma_kernel(GParams p) {
    __shared__ __align__(16) unsigned short sA[2][BM * LDK];        //  64x64 bf16 / stage
    __shared__ __align__(16) unsigned short sW[2][4 * BN * LDK];    // 128x64 bf16 / stage

    const int tid     = threadIdx.x;
    const int rowBase = blockIdx.y * BM;
    const int colBase = blockIdx.x * BN;

    // async-fill assignment: A = 2 chunks/thread, W = 4 chunks/thread (16B each)
    const int a_row  = tid >> 2;            // 0..63
    const int a_ks   = (tid & 3) << 4;      // 0/16/32/48 (bf16 elems)
    const int w_gate = tid >> 6;            // 0..3
    const int w_col  = (tid >> 1) & 31;     // 0..31
    const int w_ks   = (tid & 1) << 5;      // 0/32 (bf16 elems)

    // compute assignment: 8 waves = 4(M) x 2(N) tiles of 16x16, x4 gates
    const int wid  = tid >> 5;
    const int wm   = wid & 3;
    const int wn   = wid >> 2;
    const int lane = tid & 31;
    const int ln   = lane & 15;
    const int lh   = lane >> 4;

    const unsigned short* gA =
        p.A + (size_t)(rowBase + a_row) * K_TOTAL + a_ks;
    const unsigned short* gW =
        p.W + (size_t)(w_gate * HID_DIM + colBase + w_col) * K_TOTAL + w_ks;
    const unsigned int ldsA0 = (unsigned int)(size_t)&sA[0][a_row * LDK + a_ks];
    const unsigned int ldsA1 = (unsigned int)(size_t)&sA[1][a_row * LDK + a_ks];
    const unsigned int ldsW0 = (unsigned int)(size_t)&sW[0][(w_gate * BN + w_col) * LDK + w_ks];
    const unsigned int ldsW1 = (unsigned int)(size_t)&sW[1][(w_gate * BN + w_col) * LDK + w_ks];

    v8f acc[4];
    #pragma unroll
    for (int g = 0; g < 4; ++g) { v8f z = {}; acc[g] = z; }

    auto issue_async = [&](int s, int buf) {
        const int k0 = s * KC;                       // bf16-element offset
        const unsigned int la = buf ? ldsA1 : ldsA0;
        const unsigned int lw = buf ? ldsW1 : ldsW0;
        async_b128(la,      gA + k0);
        async_b128(la + 16, gA + k0 + 8);
        #pragma unroll
        for (int i = 0; i < 4; ++i)
            async_b128(lw + 16 * i, gW + k0 + 8 * i);
    };

    auto compute_stage = [&](int buf) {
        #pragma unroll
        for (int kk = 0; kk < KC; kk += 32) {
            // A frag: 16-bit A 16x32 layout — lane M=ln, K split by lane-half
            Frag a;
            const unsigned short* ap = &sA[buf][(wm * 16 + ln) * LDK + kk + 8 * lh];
            a.q[0] = *(const uint4*)ap;
            a.q[1] = *(const uint4*)(ap + 16);
            // B frags for all 4 gates in DISTINCT registers (batch ds_loads)
            Frag b[4];
            #pragma unroll
            for (int g = 0; g < 4; ++g) {
                const unsigned short* bp =
                    &sW[buf][(g * BN + wn * 16 + ln) * LDK + kk + 8 * lh];
                b[g].q[0] = *(const uint4*)bp;
                b[g].q[1] = *(const uint4*)(bp + 16);
            }
            #pragma unroll
            for (int g = 0; g < 4; ++g)
                acc[g] = __builtin_amdgcn_wmma_f32_16x16x32_bf16(
                    false, a.v, false, b[g].v, (short)0, acc[g], false, false);
        }
    };

    // -------- pipelined K loop: async DMA into LDS buffer s+1, WMMA on buffer s ----
    issue_async(0, 0);
    wait_async0();
    __syncthreads();
    for (int s = 0; s < NSTAGES; ++s) {
        const int buf = s & 1;
        if (s + 1 < NSTAGES) issue_async(s + 1, buf ^ 1);
        compute_stage(buf);
        wait_async0();
        __syncthreads();
    }

    // ---------------- fused LSTM epilogue ----------------
    const int j = colBase + wn * 16 + ln;
    float bias[4];
    #pragma unroll
    for (int g = 0; g < 4; ++g) bias[g] = p.bx[g][j] + p.bh[g][j];

    const int rbase = rowBase + wm * 16 + lh * 8;   // C/D layout: lanes 16-31 => M=8..15
    #pragma unroll
    for (int r = 0; r < 8; ++r) {
        const size_t off = (size_t)(rbase + r) * HID_DIM + j;
        const float iv = sigf(acc[0][r] + bias[0]);
        const float gv = tanhfast(acc[1][r] + bias[1]);
        const float fv = sigf(acc[2][r] + bias[2]);
        const float ov = sigf(acc[3][r] + bias[3]);
        const float cn = fv * p.c[off] + iv * gv;
        const float hn = ov * tanhfast(cn);
        p.h_out[off] = hn;
        p.c_out[off] = cn;
    }
}

// ==================================== launch =======================================
extern "C" void kernel_launch(void* const* d_in, const int* in_sizes, int n_in,
                              void* d_out, int out_size, void* d_ws, size_t ws_size,
                              hipStream_t stream) {
    (void)in_sizes; (void)n_in; (void)out_size; (void)ws_size;

    unsigned short* Abuf = (unsigned short*)d_ws;                               // 33.5 MB
    unsigned short* Wbuf = (unsigned short*)((char*)d_ws + (size_t)BATCH * K_TOTAL * 2); // 4 MB

    // Phase 1a: A = [x | h] in bf16
    convert_A_kernel<<<dim3((BATCH * K_TOTAL / 8) / 256), dim3(256), 0, stream>>>(
        (const float*)d_in[0], (const float*)d_in[1], Abuf);

    // Phase 1b: W = gate-major [Wx | Wh] in bf16 (gate order i, g, f, o)
    WParams wp;
    wp.Wx[0] = (const float*)d_in[3];  wp.Wh[0] = (const float*)d_in[11];  // i
    wp.Wx[1] = (const float*)d_in[9];  wp.Wh[1] = (const float*)d_in[17];  // g
    wp.Wx[2] = (const float*)d_in[7];  wp.Wh[2] = (const float*)d_in[15];  // f
    wp.Wx[3] = (const float*)d_in[5];  wp.Wh[3] = (const float*)d_in[13];  // o
    wp.out = Wbuf;
    convert_W_kernel<<<dim3((4 * HID_DIM * K_TOTAL / 8) / 256), dim3(256), 0, stream>>>(wp);

    // Phase 2: fused GEMM + LSTM
    GParams p;
    p.A = Abuf;
    p.W = Wbuf;
    p.c = (const float*)d_in[2];
    p.bx[0] = (const float*)d_in[4];  p.bh[0] = (const float*)d_in[12];  // i
    p.bx[1] = (const float*)d_in[10]; p.bh[1] = (const float*)d_in[18];  // g
    p.bx[2] = (const float*)d_in[8];  p.bh[2] = (const float*)d_in[16];  // f
    p.bx[3] = (const float*)d_in[6];  p.bh[3] = (const float*)d_in[14];  // o
    p.h_out = (float*)d_out;
    p.c_out = (float*)d_out + (size_t)BATCH * HID_DIM;

    dim3 grid(HID_DIM / BN, BATCH / BM);   // 16 x 256
    lstm_wmma_kernel<<<grid, dim3(256), 0, stream>>>(p);
}
